// JointTransformerBlock_46806553592145
// MI455X (gfx1250) — compile-verified, matrix operand order checked by reading, not run on hardware
//
#include <hip/hip_runtime.h>

// ---------------- types ----------------
typedef __attribute__((ext_vector_type(16))) __bf16 v16bf;
typedef __attribute__((ext_vector_type(8)))  float  v8f;
typedef int v4i_ __attribute__((vector_size(16)));   // matches builtin param type

#define DEV static __device__ __forceinline__

// -------- CDNA5 async global->LDS copy (guarded; falls back to ds path) -----
#if defined(__has_builtin)
#if __has_builtin(__builtin_amdgcn_global_load_async_to_lds_b128)
#define USE_ASYNC_LDS 1
#endif
#endif
#ifndef USE_ASYNC_LDS
#define USE_ASYNC_LDS 0
#endif

#if USE_ASYNC_LDS
DEV void async_cp16(const void* g, void* l) {
  __builtin_amdgcn_global_load_async_to_lds_b128(
      (__attribute__((address_space(1))) v4i_*)(g),
      (__attribute__((address_space(3))) v4i_*)(l), 0, 0);
}
DEV void wait_async0() {
#if __has_builtin(__builtin_amdgcn_s_wait_asynccnt)
  __builtin_amdgcn_s_wait_asynccnt(0);
#else
  asm volatile("s_wait_asynccnt 0" ::: "memory");
#endif
}
#endif

// D = A(16x32 bf16) * B(32x16 bf16) + C(16x16 f32)
DEV v8f wmma_bf16(v16bf a, v16bf b, v8f c) {
  return __builtin_amdgcn_wmma_f32_16x16x32_bf16(false, a, false, b, (short)0, c, false, false);
}

// Load a 16x32 fragment (A: row=M, or B: row=N) from LDS laid out row-major
// [row][k] with `stride` halfwords per row.
// lanes 0-15:  row = base+lane,    K = {k0..k0+7, k0+16..k0+23}
// lanes 16-31: row = base+lane-16, K = {k0+8..k0+15, k0+24..k0+31}
DEV v16bf ldfrag(const __bf16* lds, int row_base, int k0, int stride) {
  int lane = threadIdx.x & 31;
  const __bf16* p = lds + (size_t)(row_base + (lane & 15)) * stride + k0 + ((lane >> 4) << 3);
  union { uint4 q[2]; v16bf v; } u;
  u.q[0] = *(const uint4*)(p);
  u.q[1] = *(const uint4*)(p + 16);
  return u.v;
}

// ---------------- dims ----------------
#define Bb 2
#define SS 2048
#define DD 2048
#define HH 16
#define KVh 8
#define HD 128
#define HIDN 8192
#define CONDW 1024
#define NTOK (Bb * SS)          // 4096
#define EPSN 1e-5f
#define QKEPS 1.1920929e-07f

// =====================================================================
// 1) adaLN modulation: mod[b, j] = silu(ada[b]) . mod_w[j] + mod_b[j]
// =====================================================================
__global__ __launch_bounds__(256) void mod_kernel(const float* __restrict__ ada,
                                                  const float* __restrict__ mw,
                                                  const float* __restrict__ mb,
                                                  float* __restrict__ mod) {
  int gid = blockIdx.x * 256 + threadIdx.x;       // 0..16383
  int b = gid >> 13;
  int j = gid & 8191;
  const float* a = ada + (size_t)b * CONDW;
  const float* w = mw + (size_t)j * CONDW;
  float acc = 0.f;
  for (int c = 0; c < CONDW; c += 4) {
    float4 av = *(const float4*)(a + c);
    float4 wv = *(const float4*)(w + c);
    acc += (av.x / (1.f + __expf(-av.x))) * wv.x;
    acc += (av.y / (1.f + __expf(-av.y))) * wv.y;
    acc += (av.z / (1.f + __expf(-av.z))) * wv.z;
    acc += (av.w / (1.f + __expf(-av.w))) * wv.w;
  }
  mod[gid] = acc + mb[j];
}

// =====================================================================
// 2) h = rms(x, w, EPS) * (1 + mod[b, chunk*D + d])  -> bf16
// =====================================================================
__global__ __launch_bounds__(256) void rmsmod_kernel(const float* __restrict__ x,
                                                     const float* __restrict__ w,
                                                     const float* __restrict__ mod,
                                                     int chunk,
                                                     __bf16* __restrict__ out) {
  int tok = blockIdx.x, t = threadIdx.x;
  int b = tok >> 11;
  const float* xr = x + (size_t)tok * DD;
  __shared__ float red[256];
  float s = 0.f;
#pragma unroll
  for (int i = 0; i < 8; ++i) { float v = xr[t + i * 256]; s += v * v; }
  red[t] = s; __syncthreads();
  for (int o = 128; o > 0; o >>= 1) { if (t < o) red[t] += red[t + o]; __syncthreads(); }
  float rs = rsqrtf(red[0] * (1.f / DD) + EPSN);
  const float* mc = mod + (size_t)b * 4 * DD + (size_t)chunk * DD;
  __bf16* orow = out + (size_t)tok * DD;
#pragma unroll
  for (int i = 0; i < 8; ++i) {
    int d = t + i * 256;
    orow[d] = (__bf16)(xr[d] * rs * w[d] * (1.f + mc[d]));
  }
}

// =====================================================================
// 3) residual: out = base + tanh(mod[b,chunk*D+d]) * rms(y, w, EPS)
// =====================================================================
__global__ __launch_bounds__(256) void residual_kernel(const float* __restrict__ base,
                                                       const float* __restrict__ y,
                                                       const float* __restrict__ w,
                                                       const float* __restrict__ mod,
                                                       int chunk,
                                                       float* __restrict__ out) {
  int tok = blockIdx.x, t = threadIdx.x;
  int b = tok >> 11;
  const float* yr = y + (size_t)tok * DD;
  const float* br = base + (size_t)tok * DD;
  __shared__ float red[256];
  float s = 0.f;
#pragma unroll
  for (int i = 0; i < 8; ++i) { float v = yr[t + i * 256]; s += v * v; }
  red[t] = s; __syncthreads();
  for (int o = 128; o > 0; o >>= 1) { if (t < o) red[t] += red[t + o]; __syncthreads(); }
  float rs = rsqrtf(red[0] * (1.f / DD) + EPSN);
  const float* mc = mod + (size_t)b * 4 * DD + (size_t)chunk * DD;
  float* orow = out + (size_t)tok * DD;
#pragma unroll
  for (int i = 0; i < 8; ++i) {
    int d = t + i * 256;
    orow[d] = br[d] + tanhf(mc[d]) * (yr[d] * rs * w[d]);
  }
}

// =====================================================================
// 4) GEMM: C[M,N] f32 = A[M,K] bf16 x W[N,K] f32(->bf16). 128x128 tile.
//    Double-buffered LDS, single barrier per K-step, async A staging.
// =====================================================================
__global__ __launch_bounds__(256) void gemm_bf16_f32(const __bf16* __restrict__ A,
                                                     const float* __restrict__ W,
                                                     float* __restrict__ C,
                                                     int M, int N, int K) {
  __shared__ __align__(16) __bf16 As[2][128 * 32];
  __shared__ __align__(16) __bf16 Ws[2][128 * 32];
  int tid = threadIdx.x;
  int m0 = blockIdx.y * 128, n0 = blockIdx.x * 128;
  int wid = tid >> 5, lane = tid & 31;
  int wm = wid >> 1, wn = wid & 1;
  int hi = (lane >> 4) << 3;
  v8f acc[2][4] = {};

  auto stage = [&](int bufi, int kt) {
#pragma unroll
    for (int i = 0; i < 2; ++i) {              // A tile: 512 x 16B chunks
      int c = tid * 2 + i, row = c >> 2, part = c & 3;
      const __bf16* g = A + (size_t)(m0 + row) * K + kt + part * 8;
      __bf16* l = &As[bufi][row * 32 + part * 8];
#if USE_ASYNC_LDS
      async_cp16(g, l);
#else
      *(uint4*)l = *(const uint4*)g;
#endif
    }
#pragma unroll
    for (int i = 0; i < 4; ++i) {              // W tile: 1024 float4, cvt bf16
      int c = tid * 4 + i, row = c >> 3, part = c & 7;
      const float* gw = W + (size_t)(n0 + row) * K + kt + part * 4;
      if (i == 0 && kt + 64 < K) __builtin_prefetch(gw + 64, 0, 1);
      float4 f = *(const float4*)gw;
      __bf16* d = &Ws[bufi][row * 32 + part * 4];
      d[0] = (__bf16)f.x; d[1] = (__bf16)f.y; d[2] = (__bf16)f.z; d[3] = (__bf16)f.w;
    }
  };

  stage(0, 0);
  int nk = K >> 5;
  int buf = 0;
  for (int it = 0; it < nk; ++it) {
#if USE_ASYNC_LDS
    wait_async0();
#endif
    __syncthreads();                    // buf's tile (written last iter) ready
    if (it + 1 < nk) stage(buf ^ 1, (it + 1) << 5);   // fill other buffer
    const __bf16* Ab = As[buf];
    const __bf16* Wb = Ws[buf];
    v16bf af[2], bfg[4];
#pragma unroll
    for (int mi = 0; mi < 2; ++mi) af[mi] = ldfrag(Ab, wm * 32 + mi * 16, 0, 32);
#pragma unroll
    for (int ni = 0; ni < 4; ++ni) bfg[ni] = ldfrag(Wb, wn * 64 + ni * 16, 0, 32);
#pragma unroll
    for (int mi = 0; mi < 2; ++mi)
#pragma unroll
      for (int ni = 0; ni < 4; ++ni)
        acc[mi][ni] = wmma_bf16(af[mi], bfg[ni], acc[mi][ni]);
    buf ^= 1;
  }
#pragma unroll
  for (int mi = 0; mi < 2; ++mi)
#pragma unroll
    for (int ni = 0; ni < 4; ++ni) {
      int rbase = m0 + wm * 32 + mi * 16 + hi;
      int col = n0 + wn * 64 + ni * 16 + (lane & 15);
#pragma unroll
      for (int v = 0; v < 8; ++v)
        C[(size_t)(rbase + v) * N + col] = acc[mi][ni][v];
    }
}

// =====================================================================
// 5) Fused SwiGLU up-projection: out = silu(A@W1^T) * (A@W3^T) -> bf16
//    Same double-buffered single-barrier pipeline.
// =====================================================================
__global__ __launch_bounds__(256) void gemm_ffn13(const __bf16* __restrict__ A,
                                                  const float* __restrict__ W1,
                                                  const float* __restrict__ W3,
                                                  __bf16* __restrict__ Cb,
                                                  int M, int N, int K) {
  __shared__ __align__(16) __bf16 As[2][128 * 32];
  __shared__ __align__(16) __bf16 W1s[2][128 * 32];
  __shared__ __align__(16) __bf16 W3s[2][128 * 32];
  int tid = threadIdx.x;
  int m0 = blockIdx.y * 128, n0 = blockIdx.x * 128;
  int wid = tid >> 5, lane = tid & 31;
  int wm = wid >> 1, wn = wid & 1;
  int hi = (lane >> 4) << 3;
  v8f acc1[2][4] = {}, acc3[2][4] = {};

  auto stage = [&](int bufi, int kt) {
#pragma unroll
    for (int i = 0; i < 2; ++i) {
      int c = tid * 2 + i, row = c >> 2, part = c & 3;
      const __bf16* g = A + (size_t)(m0 + row) * K + kt + part * 8;
      __bf16* l = &As[bufi][row * 32 + part * 8];
#if USE_ASYNC_LDS
      async_cp16(g, l);
#else
      *(uint4*)l = *(const uint4*)g;
#endif
    }
#pragma unroll
    for (int i = 0; i < 4; ++i) {
      int c = tid * 4 + i, row = c >> 3, part = c & 7;
      const float* g1 = W1 + (size_t)(n0 + row) * K + kt + part * 4;
      const float* g3 = W3 + (size_t)(n0 + row) * K + kt + part * 4;
      if (i == 0 && kt + 64 < K) {
        __builtin_prefetch(g1 + 64, 0, 1);
        __builtin_prefetch(g3 + 64, 0, 1);
      }
      float4 f1 = *(const float4*)g1;
      float4 f3 = *(const float4*)g3;
      __bf16* d1 = &W1s[bufi][row * 32 + part * 4];
      __bf16* d3 = &W3s[bufi][row * 32 + part * 4];
      d1[0] = (__bf16)f1.x; d1[1] = (__bf16)f1.y; d1[2] = (__bf16)f1.z; d1[3] = (__bf16)f1.w;
      d3[0] = (__bf16)f3.x; d3[1] = (__bf16)f3.y; d3[2] = (__bf16)f3.z; d3[3] = (__bf16)f3.w;
    }
  };

  stage(0, 0);
  int nk = K >> 5;
  int buf = 0;
  for (int it = 0; it < nk; ++it) {
#if USE_ASYNC_LDS
    wait_async0();
#endif
    __syncthreads();
    if (it + 1 < nk) stage(buf ^ 1, (it + 1) << 5);
    const __bf16* Ab = As[buf];
    const __bf16* B1 = W1s[buf];
    const __bf16* B3 = W3s[buf];
    v16bf af[2], b1[4], b3[4];
#pragma unroll
    for (int mi = 0; mi < 2; ++mi) af[mi] = ldfrag(Ab, wm * 32 + mi * 16, 0, 32);
#pragma unroll
    for (int ni = 0; ni < 4; ++ni) {
      b1[ni] = ldfrag(B1, wn * 64 + ni * 16, 0, 32);
      b3[ni] = ldfrag(B3, wn * 64 + ni * 16, 0, 32);
    }
#pragma unroll
    for (int mi = 0; mi < 2; ++mi)
#pragma unroll
      for (int ni = 0; ni < 4; ++ni) {
        acc1[mi][ni] = wmma_bf16(af[mi], b1[ni], acc1[mi][ni]);
        acc3[mi][ni] = wmma_bf16(af[mi], b3[ni], acc3[mi][ni]);
      }
    buf ^= 1;
  }
#pragma unroll
  for (int mi = 0; mi < 2; ++mi)
#pragma unroll
    for (int ni = 0; ni < 4; ++ni) {
      int rbase = m0 + wm * 32 + mi * 16 + hi;
      int col = n0 + wn * 64 + ni * 16 + (lane & 15);
#pragma unroll
      for (int v = 0; v < 8; ++v) {
        float v1 = acc1[mi][ni][v];
        float v3 = acc3[mi][ni][v];
        float r = (v1 / (1.f + __expf(-v1))) * v3;
        Cb[(size_t)(rbase + v) * N + col] = (__bf16)r;
      }
    }
}

// =====================================================================
// 6) QKV post: per (token, head) RMSNorm(q/k) + RoPE, v passthrough -> bf16
// =====================================================================
__global__ __launch_bounds__(128) void qkv_post_kernel(const float* __restrict__ qkv,
                                                       const float* __restrict__ freqs,
                                                       const float* __restrict__ qw,
                                                       const float* __restrict__ kw,
                                                       __bf16* __restrict__ Q,
                                                       __bf16* __restrict__ Kk,
                                                       __bf16* __restrict__ V) {
  int tok = blockIdx.x, head = blockIdx.y, t = threadIdx.x;
  int b = tok >> 11, s = tok & 2047;
  float val = qkv[(size_t)tok * 4096 + head * HD + t];
  if (head >= HH + KVh) {                 // V: convert only (uniform per block)
    V[(((size_t)b * KVh + (head - HH - KVh)) * SS + s) * HD + t] = (__bf16)val;
    return;
  }
  __shared__ float red[128];
  __shared__ float arr[128];
  red[t] = val * val; __syncthreads();
  for (int o = 64; o > 0; o >>= 1) { if (t < o) red[t] += red[t + o]; __syncthreads(); }
  float rs = rsqrtf(red[0] * (1.f / HD) + QKEPS);
  const float* w = (head < HH) ? qw : kw;
  arr[t] = val * rs * w[t];
  __syncthreads();
  if (t < 64) {
    float x0 = arr[2 * t], x1 = arr[2 * t + 1];
    const float* fc = freqs + (size_t)s * 256 + t * 4;   // [S,1,64,2,2]
    float o0 = fc[0] * x0 + fc[1] * x1;
    float o1 = fc[2] * x0 + fc[3] * x1;
    if (head < HH) {
      size_t base = (((size_t)b * HH + head) * SS + s) * HD;
      Q[base + 2 * t] = (__bf16)o0; Q[base + 2 * t + 1] = (__bf16)o1;
    } else {
      size_t base = (((size_t)b * KVh + (head - HH)) * SS + s) * HD;
      Kk[base + 2 * t] = (__bf16)o0; Kk[base + 2 * t + 1] = (__bf16)o1;
    }
  }
}

// =====================================================================
// 7) Flash attention: block = (b, h, 64 q-rows), 64-key tiles.
// =====================================================================
__global__ __launch_bounds__(256) void attn_kernel(const __bf16* __restrict__ Qg,
                                                   const __bf16* __restrict__ Kg,
                                                   const __bf16* __restrict__ Vg,
                                                   __bf16* __restrict__ O) {
  extern __shared__ __align__(16) char smem[];
  __bf16* Qs = (__bf16*)smem;                 // 64 x 128
  __bf16* Ks = Qs + 64 * 128;                 // 64 x 128
  __bf16* Vt = Ks + 64 * 128;                 // 128(hd) x 64(key)
  float*  Sf = (float*)(Vt + 128 * 64);       // 64 x 64 f32
  __bf16* Ps = (__bf16*)(Sf + 64 * 64);       // 64 x 64
  float*  rowc = (float*)(Ps + 64 * 64);      // 64 f32

  int tid = threadIdx.x, lane = tid & 31, wid = tid >> 5;
  int hi = (lane >> 4) << 3;
  int q0 = blockIdx.x * 64, h = blockIdx.y, b = blockIdx.z;
  int kvh = h >> 1;                            // H/KV == 2
  const __bf16* Qbase = Qg + (((size_t)b * HH + h) * SS + q0) * HD;
  const __bf16* Kbase = Kg + (((size_t)b * KVh + kvh) * SS) * HD;
  const __bf16* Vbase = Vg + (((size_t)b * KVh + kvh) * SS) * HD;

#pragma unroll
  for (int i = 0; i < 4; ++i) {               // stage Q once: 1024 x 16B
    int c = tid * 4 + i, row = c >> 4, part = c & 15;
    const __bf16* g = Qbase + (size_t)row * HD + part * 8;
    __bf16* l = Qs + row * 128 + part * 8;
#if USE_ASYNC_LDS
    async_cp16(g, l);
#else
    *(uint4*)l = *(const uint4*)g;
#endif
  }

  float m_r = -1e30f, l_r = 0.f;
  int swm = wid & 3, swn = wid >> 2;          // S-gemm wave grid: 4(M) x 2
  int owm = wid & 1, own = wid >> 1;          // O-gemm wave grid: 2(M) x 4
  v8f o_acc[2][2] = {};
  const float sc = 0.08838834764831845f;      // 1/sqrt(128)

  for (int j = 0; j < SS; j += 64) {
    __syncthreads();
#pragma unroll
    for (int i = 0; i < 4; ++i) {             // stage K tile
      int c = tid * 4 + i, row = c >> 4, part = c & 15;
      const __bf16* g = Kbase + (size_t)(j + row) * HD + part * 8;
      __bf16* l = Ks + row * 128 + part * 8;
#if USE_ASYNC_LDS
      async_cp16(g, l);
#else
      *(uint4*)l = *(const uint4*)g;
#endif
    }
#pragma unroll
    for (int i = 0; i < 32; ++i) {            // stage V transposed: [hd][key]
      int e = i * 256 + tid;
      int hd = e & 127, key = e >> 7;
      Vt[hd * 64 + key] = Vbase[(size_t)(j + key) * HD + hd];
    }
#if USE_ASYNC_LDS
    wait_async0();
#endif
    __syncthreads();

    // S = Q K^T
    v8f s_acc[2] = {};
#pragma unroll
    for (int kt = 0; kt < 128; kt += 32) {
      v16bf aq = ldfrag(Qs, swm * 16, kt, 128);
#pragma unroll
      for (int ni = 0; ni < 2; ++ni) {
        v16bf bk = ldfrag(Ks, swn * 32 + ni * 16, kt, 128);
        s_acc[ni] = wmma_bf16(aq, bk, s_acc[ni]);
      }
    }
#pragma unroll
    for (int ni = 0; ni < 2; ++ni)
#pragma unroll
      for (int v = 0; v < 8; ++v) {
        int row = swm * 16 + hi + v;
        int col = swn * 32 + ni * 16 + (lane & 15);
        Sf[row * 64 + col] = s_acc[ni][v] * sc;
      }
    __syncthreads();

    // online softmax: thread r owns row r
    if (tid < 64) {
      const float* srow = Sf + tid * 64;
      float rm = srow[0];
      for (int c = 1; c < 64; ++c) rm = fmaxf(rm, srow[c]);
      float mnew = fmaxf(m_r, rm);
      float corr = __expf(m_r - mnew);
      float sum = 0.f;
      __bf16* prow = Ps + tid * 64;
      for (int c = 0; c < 64; ++c) {
        float p = __expf(srow[c] - mnew);
        sum += p;
        prow[c] = (__bf16)p;
      }
      l_r = l_r * corr + sum;
      m_r = mnew;
      rowc[tid] = corr;
    }
    __syncthreads();

    // rescale O accumulator then O += P V
#pragma unroll
    for (int mi = 0; mi < 2; ++mi)
#pragma unroll
      for (int ni = 0; ni < 2; ++ni) {
        int rbase = owm * 32 + mi * 16 + hi;
#pragma unroll
        for (int v = 0; v < 8; ++v) o_acc[mi][ni][v] *= rowc[rbase + v];
      }
#pragma unroll
    for (int kt = 0; kt < 64; kt += 32) {
      v16bf ap[2], bv[2];
#pragma unroll
      for (int mi = 0; mi < 2; ++mi) ap[mi] = ldfrag(Ps, owm * 32 + mi * 16, kt, 64);
#pragma unroll
      for (int ni = 0; ni < 2; ++ni) bv[ni] = ldfrag(Vt, own * 32 + ni * 16, kt, 64);
#pragma unroll
      for (int mi = 0; mi < 2; ++mi)
#pragma unroll
        for (int ni = 0; ni < 2; ++ni)
          o_acc[mi][ni] = wmma_bf16(ap[mi], bv[ni], o_acc[mi][ni]);
    }
  }
  __syncthreads();
  if (tid < 64) rowc[tid] = 1.f / l_r;
  __syncthreads();
#pragma unroll
  for (int mi = 0; mi < 2; ++mi)
#pragma unroll
    for (int ni = 0; ni < 2; ++ni) {
      int rbase = owm * 32 + mi * 16 + hi;
      int col = own * 32 + ni * 16 + (lane & 15);
#pragma unroll
      for (int v = 0; v < 8; ++v) {
        int row = rbase + v;
        float val = o_acc[mi][ni][v] * rowc[row];
        O[(((size_t)b * SS + q0 + row) * HH + h) * HD + col] = (__bf16)val;
      }
    }
}

// =====================================================================
// launch
// =====================================================================
extern "C" void kernel_launch(void* const* d_in, const int* in_sizes, int n_in,
                              void* d_out, int out_size, void* d_ws, size_t ws_size,
                              hipStream_t stream) {
  (void)in_sizes; (void)n_in; (void)out_size; (void)ws_size;
  const float* x      = (const float*)d_in[0];
  const float* freqs  = (const float*)d_in[2];
  const float* ada    = (const float*)d_in[3];
  const float* mod_w  = (const float*)d_in[4];
  const float* mod_b  = (const float*)d_in[5];
  const float* qkv_w  = (const float*)d_in[6];
  const float* out_w  = (const float*)d_in[7];
  const float* q_nw   = (const float*)d_in[8];
  const float* k_nw   = (const float*)d_in[9];
  const float* an1    = (const float*)d_in[10];
  const float* an2    = (const float*)d_in[11];
  const float* fn1    = (const float*)d_in[12];
  const float* fn2    = (const float*)d_in[13];
  const float* w1     = (const float*)d_in[14];
  const float* w2     = (const float*)d_in[15];
  const float* w3     = (const float*)d_in[16];
  float* out = (float*)d_out;

  const size_t MBy = 1ull << 20;
  char* ws = (char*)d_ws;
  float*  modBuf = (float*)(ws + 0);             //  64 KB
  __bf16* hA     = (__bf16*)(ws + 1 * MBy);      //  16 MB (reused as h2)
  float*  qkvBuf = (float*)(ws + 17 * MBy);      //  64 MB
  __bf16* ffh    = (__bf16*)(ws + 17 * MBy);     //  (reuse: ffh bf16, 64 MB)
  __bf16* Qb     = (__bf16*)(ws + 81 * MBy);     //  16 MB
  __bf16* Kb     = (__bf16*)(ws + 97 * MBy);     //   8 MB
  __bf16* Vb     = (__bf16*)(ws + 105 * MBy);    //   8 MB
  __bf16* attnO  = (__bf16*)(ws + 113 * MBy);    //  16 MB
  float*  oproj  = (float*)(ws + 129 * MBy);     //  32 MB (reused as ff)
  float*  ffBuf  = (float*)(ws + 129 * MBy);
  float*  x1     = (float*)(ws + 161 * MBy);     //  32 MB  -> 193 MB total

  mod_kernel<<<64, 256, 0, stream>>>(ada, mod_w, mod_b, modBuf);
  rmsmod_kernel<<<NTOK, 256, 0, stream>>>(x, an1, modBuf, 0, hA);
  gemm_bf16_f32<<<dim3(4096 / 128, NTOK / 128), 256, 0, stream>>>(hA, qkv_w, qkvBuf,
                                                                  NTOK, 4096, DD);
  qkv_post_kernel<<<dim3(NTOK, 32), 128, 0, stream>>>(qkvBuf, freqs, q_nw, k_nw, Qb, Kb, Vb);
  size_t smem = (size_t)(64 * 128 + 64 * 128 + 128 * 64) * 2 + 64 * 64 * 4 + 64 * 64 * 2 + 64 * 4;
  attn_kernel<<<dim3(SS / 64, HH, Bb), 256, smem, stream>>>(Qb, Kb, Vb, attnO);
  gemm_bf16_f32<<<dim3(DD / 128, NTOK / 128), 256, 0, stream>>>(attnO, out_w, oproj,
                                                                NTOK, DD, DD);
  residual_kernel<<<NTOK, 256, 0, stream>>>(x, oproj, an2, modBuf, 1, x1);
  rmsmod_kernel<<<NTOK, 256, 0, stream>>>(x1, fn1, modBuf, 2, hA);
  gemm_ffn13<<<dim3(HIDN / 128, NTOK / 128), 256, 0, stream>>>(hA, w1, w3, ffh,
                                                               NTOK, HIDN, DD);
  gemm_bf16_f32<<<dim3(DD / 128, NTOK / 128), 256, 0, stream>>>(ffh, w2, ffBuf,
                                                                NTOK, DD, HIDN);
  residual_kernel<<<NTOK, 256, 0, stream>>>(x1, ffBuf, fn2, modBuf, 3, out);
}
